// HadamardAdapter_70987219468450
// MI455X (gfx1250) — matrix operand order repaired and checked by reading, not compiled, not run on hardware
//
#include <hip/hip_runtime.h>

typedef __attribute__((ext_vector_type(16))) _Float16 v16h;
typedef __attribute__((ext_vector_type(8)))  float    v8f;

#define DIM        4096
#define NROWS      8192            // B*T = 4*2048
#define KCOMP      24
#define NSPLIT     4               // K-dimension split for occupancy
#define KCHUNK     (DIM / NSPLIT)  // 1024 (multiple of 32 -> B invariance holds)
#define NTILES     (NROWS / 16)    // 512 row-tiles
#define NWAVES     (NTILES * NSPLIT) // 2048 phase-1 waves

// workspace layout (float offsets)
#define WS_Z     0                               // NSPLIT * NROWS * 32  (4 MB)
#define WS_PART  (NSPLIT * NROWS * 32)           // NWAVES * 32
#define WS_COEF  (WS_PART + NWAVES * 32)         // 32
#define WS_MEAN  (WS_COEF + 32)                  // 32

// ---------------------------------------------------------------------------
// Phase 1: partial z = hs @ H[0:24,:].T via WMMA f32<-f16 with hi/lo split
// (fp32-accurate: A = A_hi + A_lo, H entries +-1/64 exact in f16).
// One wave per (16-row tile, K-quarter); 32 chunk iterations of K=32.
// B fragment is loop-invariant: H[j][d] = (-1)^popc(j&d)/64 depends on d&31
// only (j < 32), and kbase stays 32-aligned.
// ---------------------------------------------------------------------------
__global__ __launch_bounds__(32)
void had_z_wmma(const float* __restrict__ hs, float* __restrict__ ws) {
    const int lane  = threadIdx.x;
    const int m     = lane & 15;
    const int h     = lane >> 4;
    const int bid   = blockIdx.x;          // 0..NWAVES-1
    const int tile  = bid >> 2;            // row tile
    const int ks    = bid & (NSPLIT - 1);  // K quarter
    const int rbase = tile * 16;
    const int kstart = ks * KCHUNK;
    const float* rowp = hs + (size_t)(rbase + m) * DIM;

    // Constant B fragments. Lane layout (16-bit B 32x16): N = lane&15,
    // element e (VGPR e/2, half e&1) holds K = 8h + e (e<8), 16 + 8h + e-8 (e>=8).
    v16h b0, b1;
    const _Float16 pos = (_Float16)( 1.0f / 64.0f);
    const _Float16 neg = (_Float16)(-1.0f / 64.0f);
    #pragma unroll
    for (int e = 0; e < 16; ++e) {
        const int kb = (e < 8) ? (8 * h + e) : (16 + 8 * h + (e - 8));          // == d & 31
        const int j0 = m;                    // cols 0..15
        const int j1 = 16 + m;               // cols 16..31 (>=24 zeroed)
        b0[e] = (__builtin_popcount(j0 & kb) & 1) ? neg : pos;
        _Float16 v1 = (__builtin_popcount(j1 & kb) & 1) ? neg : pos;
        b1[e] = (j1 < KCOMP) ? v1 : (_Float16)0.0f;
    }

    v8f c0 = {};
    v8f c1 = {};

    for (int kbase = kstart; kbase < kstart + KCHUNK; kbase += 32) {
        // A layout (16-bit A 16x32): M = lane&15; element e holds
        //   e<8 : K = 8h + e        -> contiguous group 1
        //   e>=8: K = 16 + 8h + e-8 -> contiguous group 2
        const float* p1 = rowp + kbase + 8 * h;
        const float* p2 = rowp + kbase + 16 + 8 * h;
        const float4 xa = *(const float4*)(p1);
        const float4 xb = *(const float4*)(p1 + 4);
        const float4 xc = *(const float4*)(p2);
        const float4 xd = *(const float4*)(p2 + 4);
        const float xs[16] = { xa.x, xa.y, xa.z, xa.w,  xb.x, xb.y, xb.z, xb.w,
                               xc.x, xc.y, xc.z, xc.w,  xd.x, xd.y, xd.z, xd.w };
        v16h ahi, alo;
        #pragma unroll
        for (int e = 0; e < 16; ++e) {
            const float    x  = xs[e];
            const _Float16 hi = (_Float16)x;
            ahi[e] = hi;
            alo[e] = (_Float16)(x - (float)hi);   // residual: hi+lo ~ fp32 exact
        }
        c0 = __builtin_amdgcn_wmma_f32_16x16x32_f16(false, ahi, false, b0, (short)0, c0, false, false);
        c0 = __builtin_amdgcn_wmma_f32_16x16x32_f16(false, alo, false, b0, (short)0, c0, false, false);
        c1 = __builtin_amdgcn_wmma_f32_16x16x32_f16(false, ahi, false, b1, (short)0, c1, false, false);
        c1 = __builtin_amdgcn_wmma_f32_16x16x32_f16(false, alo, false, b1, (short)0, c1, false, false);
    }

    // C/D layout: element r -> row M = r + 8h, col N = lane&15
    float* Z = ws + WS_Z + (size_t)ks * NROWS * 32;
    #pragma unroll
    for (int r = 0; r < 8; ++r) {
        const int row = rbase + 8 * h + r;
        Z[row * 32 + m]      = c0[r];
        Z[row * 32 + 16 + m] = c1[r];
    }

    // deterministic per-wave column partial sums (for the mean)
    float s0 = 0.f, s1 = 0.f;
    #pragma unroll
    for (int r = 0; r < 8; ++r) { s0 += c0[r]; s1 += c1[r]; }
    s0 += __shfl_xor(s0, 16, 32);
    s1 += __shfl_xor(s1, 16, 32);
    if (h == 0) {
        float* P = ws + WS_PART + bid * 32;
        P[m]      = s0;
        P[16 + m] = s1;
    }
}

// ---------------------------------------------------------------------------
// Phase 2: fixed-order reduction of per-wave partials -> mean; leaky_relu coeffs
// ---------------------------------------------------------------------------
__global__ __launch_bounds__(32)
void had_reduce(const float* __restrict__ adapter, float* __restrict__ ws) {
    const int j = threadIdx.x;
    if (j >= KCOMP) return;
    const float* P = ws + WS_PART;
    float sum = 0.f;
    for (int w = 0; w < NWAVES; ++w) sum += P[w * 32 + j];
    const float a = adapter[j];
    ws[WS_COEF + j] = (a > 0.f) ? a : 0.01f * a;       // jax.nn.leaky_relu default
    ws[WS_MEAN + j] = sum * (1.0f / (float)NROWS);
}

// ---------------------------------------------------------------------------
// Phase 3: out = hs + rank-24 update. For j<24 the Hadamard row pattern has
// period 32 in d, so per row we build a 32-entry LUT and stream-add.
// z for a row = deterministic sum of the NSPLIT K-partials.
// ---------------------------------------------------------------------------
__global__ __launch_bounds__(256)
void had_apply(const float* __restrict__ hs, const float* __restrict__ ws,
               float* __restrict__ out) {
    __shared__ float S[32];
    const int row = blockIdx.x;
    const int t   = threadIdx.x;
    if (t < 32) {
        const float* C  = ws + WS_COEF;
        const float* M  = ws + WS_MEAN;
        float s = 0.f;
        #pragma unroll
        for (int j = 0; j < KCOMP; ++j) {
            float z = 0.f;
            #pragma unroll
            for (int ks = 0; ks < NSPLIT; ++ks)
                z += ws[WS_Z + (size_t)ks * NROWS * 32 + (size_t)row * 32 + j];
            const float w = C[j] * (z - M[j]) * (1.0f / 64.0f);
            s += (__builtin_popcount(j & t) & 1) ? -w : w;
        }
        S[t] = s;
    }
    __syncthreads();
    const float4* src = (const float4*)(hs  + (size_t)row * DIM);
    float4*       dst = (float4*)      (out + (size_t)row * DIM);
    #pragma unroll
    for (int it = 0; it < 4; ++it) {
        const int idx4 = it * 256 + t;        // 1024 float4 per row
        float4 x = src[idx4];
        const int mm = (idx4 * 4) & 31;       // mm in {0,4,...,28}; mm+3 <= 31
        x.x += S[mm];
        x.y += S[mm + 1];
        x.z += S[mm + 2];
        x.w += S[mm + 3];
        dst[idx4] = x;
    }
}

extern "C" void kernel_launch(void* const* d_in, const int* in_sizes, int n_in,
                              void* d_out, int out_size, void* d_ws, size_t ws_size,
                              hipStream_t stream) {
    const float* hs      = (const float*)d_in[0];   // (4,2048,4096) f32
    const float* adapter = (const float*)d_in[1];   // (24,) f32
    // d_in[2] = H is never read: regenerated via (-1)^popc(j&d)/64 (Sylvester)
    float* ws  = (float*)d_ws;                      // ~4.3 MB used
    float* out = (float*)d_out;

    had_z_wmma<<<NWAVES, 32, 0, stream>>>(hs, ws);
    had_reduce<<<1, 32, 0, stream>>>(adapter, ws);
    had_apply<<<NROWS, 256, 0, stream>>>(hs, ws, out);
}
